// quantLlamaMLP_30906584662311
// MI455X (gfx1250) — compile-verified, matrix operand order checked by reading, not run on hardware
//
#include <hip/hip_runtime.h>
#include <stdint.h>

// MI455X / gfx1250, wave32. Fused quantized-Llama-MLP.
// HBM-bound (~1.2 GB traffic vs 2.2 TFLOP -> ~50us floor at 23.3 TB/s).
// Weights read exactly once; x / intermediate reuse via LDS + 192MB L2.
// Kernel 1: reg-pipelined f32->f16 staging; Kernel 2: async-to-LDS pipeline.

typedef _Float16 v16h __attribute__((ext_vector_type(16)));
typedef _Float16 v8h  __attribute__((ext_vector_type(8)));
typedef float    v8f  __attribute__((ext_vector_type(8)));

#define Mz 8192      // B*S = 4*2048
#define Hz 4096
#define Iz 11008

#define BM 64        // M rows per block
#define BN 128       // output channels per block (8 waves x 16)
#define BK 64        // K slice per staging iteration (2 WMMA sub-steps)
#define LDSS 72      // padded LDS row stride in halves (144B -> conflict-free)

// ---- ASYNCcnt wait ---------------------------------------------------------
#if __has_builtin(__builtin_amdgcn_s_wait_asynccnt)
#define WAIT_ASYNC(n) __builtin_amdgcn_s_wait_asynccnt(n)
#else
#define WAIT_ASYNC(n) asm volatile("s_wait_asynccnt %0" :: "i"(n) : "memory")
#endif

// global -> LDS async copy, 16B (VDST = LDS byte address, VADDR = 64b global)
__device__ __forceinline__ void async_ld_b128(uint32_t lds_addr, const void* gptr) {
    asm volatile("global_load_async_to_lds_b128 %0, %1, off"
                 :: "v"(lds_addr), "v"((uint64_t)(uintptr_t)gptr)
                 : "memory");
}

// A/B operand fetch for v_wmma_f32_16x16x32_f16, per CDNA5 ISA 16-bit 16x32 layout:
//   lanes 0-15  : row = lane,     K chunks {0..7}  and {16..23}
//   lanes 16-31 : row = lane-16,  K chunks {8..15} and {24..31}
// caller passes koff = (lane>=16 ? 8 : 0); chunks live at koff and koff+16.
__device__ __forceinline__ v16h lds_fetch_a(const _Float16* lds, int row, int koff) {
    const v8h* p0 = (const v8h*)(lds + row * LDSS + koff);
    const v8h* p1 = (const v8h*)(lds + row * LDSS + 16 + koff);
    v8h lo = *p0;
    v8h hi = *p1;
    v16h r;
#pragma unroll
    for (int q = 0; q < 8; ++q) { r[q] = lo[q]; r[q + 8] = hi[q]; }
    return r;
}

// weight row is int32 (int8 values); int8 is exactly representable in f16.
__device__ __forceinline__ v16h load_wq_f16(const int* __restrict__ w, int koff) {
    v16h r;
#pragma unroll
    for (int q = 0; q < 8; ++q) {
        r[q]     = (_Float16)w[koff + q];
        r[q + 8] = (_Float16)w[koff + 16 + q];
    }
    return r;
}

// ---------------- Kernel 1: gate/up projections + silu, fused ----------------
__global__ __launch_bounds__(256)
void mlp_gateup_silu(const float* __restrict__ x,
                     const int*   __restrict__ gate_wq,
                     const int*   __restrict__ up_wq,
                     const float* __restrict__ gate_scale,
                     const float* __restrict__ up_scale,
                     _Float16*    __restrict__ inter)
{
    __shared__ _Float16 xs[2][BM * LDSS];

    const int tid   = threadIdx.x;
    const int wave  = tid >> 5;
    const int lane  = tid & 31;
    const int laneN = lane & 15;
    const int hiSel = (lane >> 4) & 1;
    const int koff  = hiSel * 8;

    const int mBlock = blockIdx.x * BM;
    const int iCh    = blockIdx.y * BN + wave * 16 + laneN;

    const int* gRow = gate_wq + (size_t)iCh * Hz;
    const int* uRow = up_wq   + (size_t)iCh * Hz;

    v8f zero = {};
    v8f accG[4], accU[4];
#pragma unroll
    for (int t = 0; t < 4; ++t) { accG[t] = zero; accU[t] = zero; }

    // staging: 64x64 tile, 16 consecutive f32 per thread (coalesced 64B)
    const int xr = tid >> 2;          // 0..63
    const int xc = (tid & 3) * 16;    // 0,16,32,48

    const int nIter = Hz / BK;        // 64

    float xreg[16];
    {
        const float* xp = x + (size_t)(mBlock + xr) * Hz + xc;
#pragma unroll
        for (int q = 0; q < 16; ++q) xreg[q] = xp[q];
    }

    for (int it = 0; it < nIter; ++it) {
        const int k = it * BK;
        _Float16* buf = xs[it & 1];

        if (it) __syncthreads();           // buffer-reuse protection
        {
            v8h t0, t1;
#pragma unroll
            for (int q = 0; q < 8; ++q) { t0[q] = (_Float16)xreg[q];
                                          t1[q] = (_Float16)xreg[8 + q]; }
            *(v8h*)(buf + xr * LDSS + xc)     = t0;
            *(v8h*)(buf + xr * LDSS + xc + 8) = t1;
        }
        if (it + 1 < nIter) {              // next x tile -> regs, overlaps WMMAs
            const float* xp = x + (size_t)(mBlock + xr) * Hz + (k + BK) + xc;
#pragma unroll
            for (int q = 0; q < 16; ++q) xreg[q] = xp[q];
            __builtin_prefetch(gRow + k + BK, 0, 3);   // global_prefetch_b8 (near)
            __builtin_prefetch(uRow + k + BK, 0, 3);
        }
        __syncthreads();                   // publish tile

#pragma unroll
        for (int ks = 0; ks < BK; ks += 32) {
            v16h bG = load_wq_f16(gRow + k + ks, koff);
            v16h bU = load_wq_f16(uRow + k + ks, koff);
#pragma unroll
            for (int t = 0; t < 4; ++t) {
                v16h a = lds_fetch_a(buf + ks, t * 16 + laneN, koff);
                accG[t] = __builtin_amdgcn_wmma_f32_16x16x32_f16(
                    false, a, false, bG, (short)0, accG[t], false, false);
                accU[t] = __builtin_amdgcn_wmma_f32_16x16x32_f16(
                    false, a, false, bU, (short)0, accU[t], false, false);
            }
        }
    }

    const float gs = gate_scale[iCh];
    const float us = up_scale[iCh];

    // C layout: VGPR j -> m = tile + j + 8*hiSel, n = laneN
#pragma unroll
    for (int t = 0; t < 4; ++t) {
#pragma unroll
        for (int j = 0; j < 8; ++j) {
            float g = accG[t][j] * gs;
            float u = accU[t][j] * us;
            float s = g / (1.0f + __expf(-g));     // silu
            int   m = mBlock + t * 16 + hiSel * 8 + j;
            inter[(size_t)m * Iz + iCh] = (_Float16)(s * u);
        }
    }
}

// ---------------- Kernel 2: down projection (async-to-LDS pipeline) ----------
__global__ __launch_bounds__(256)
void mlp_down(const _Float16* __restrict__ inter,
              const int*      __restrict__ down_wq,
              const float*    __restrict__ down_scale,
              float*          __restrict__ out)
{
    __shared__ _Float16 as_[2][BM * LDSS];

    const int tid   = threadIdx.x;
    const int wave  = tid >> 5;
    const int lane  = tid & 31;
    const int laneN = lane & 15;
    const int hiSel = (lane >> 4) & 1;
    const int koff  = hiSel * 8;

    const int mBlock = blockIdx.x * BM;
    const int hCh    = blockIdx.y * BN + wave * 16 + laneN;

    const int* wRow = down_wq + (size_t)hCh * Iz;

    v8f zero = {};
    v8f acc[4];
#pragma unroll
    for (int t = 0; t < 4; ++t) acc[t] = zero;

    // staging: 64x64 halves, 16 halves (32B = 2x b128 async) per thread
    const int xr = tid >> 2;          // 0..63
    const int xc = (tid & 3) * 16;    // 0,16,32,48 (halves)

    const int nIter = Iz / BK;        // 172

    // issue both async b128 copies for tile `it` into buffer it&1
    auto issue = [&](int it) {
        const _Float16* gp = inter + (size_t)(mBlock + xr) * Iz + it * BK + xc;
        uint32_t l = (uint32_t)(uintptr_t)&as_[it & 1][xr * LDSS + xc];
        async_ld_b128(l,      gp);       // halves [xc, xc+8)
        async_ld_b128(l + 16, gp + 8);   // halves [xc+8, xc+16)
    };

    issue(0);

    for (int it = 0; it < nIter; ++it) {
        const int k = it * BK;
        const _Float16* buf = as_[it & 1];

        if (it + 1 < nIter) {
            issue(it + 1);        // tile it+1 streams into other buffer...
            WAIT_ASYNC(2);        // ...while we require tile it complete (in-order)
        } else {
            WAIT_ASYNC(0);
        }
        __syncthreads();          // publish tile it across all 8 waves

        if (k + BK < Iz) __builtin_prefetch(wRow + k + BK, 0, 3);

#pragma unroll
        for (int ks = 0; ks < BK; ks += 32) {
            v16h b = load_wq_f16(wRow + k + ks, koff);
#pragma unroll
            for (int t = 0; t < 4; ++t) {
                v16h a = lds_fetch_a(buf + ks, t * 16 + laneN, koff);
                acc[t] = __builtin_amdgcn_wmma_f32_16x16x32_f16(
                    false, a, false, b, (short)0, acc[t], false, false);
            }
        }
        __syncthreads();          // all reads of buf done before it is re-issued
    }

    const float dsc = down_scale[hCh];

#pragma unroll
    for (int t = 0; t < 4; ++t) {
#pragma unroll
        for (int j = 0; j < 8; ++j) {
            int m = mBlock + t * 16 + hiSel * 8 + j;
            out[(size_t)m * Hz + hCh] = acc[t][j] * dsc;
        }
    }
}

extern "C" void kernel_launch(void* const* d_in, const int* in_sizes, int n_in,
                              void* d_out, int out_size, void* d_ws, size_t ws_size,
                              hipStream_t stream) {
    const float* x          = (const float*)d_in[0];
    const int*   gate_wq    = (const int*)  d_in[1];
    const int*   up_wq      = (const int*)  d_in[2];
    const int*   down_wq    = (const int*)  d_in[3];
    const float* gate_scale = (const float*)d_in[4];
    const float* up_scale   = (const float*)d_in[5];
    const float* down_scale = (const float*)d_in[6];
    float*       out        = (float*)d_out;
    _Float16*    inter      = (_Float16*)d_ws;   // [M, I] f16 = 180 MB scratch

    dim3 block(256);
    dim3 grid1(Mz / BM, Iz / BN);   // 128 x 86
    dim3 grid2(Mz / BM, Hz / BN);   // 128 x 32

    mlp_gateup_silu<<<grid1, block, 0, stream>>>(x, gate_wq, up_wq,
                                                 gate_scale, up_scale, inter);
    mlp_down<<<grid2, block, 0, stream>>>(inter, down_wq, down_scale, out);
}
